// VLADLayer_65704409694209
// MI455X (gfx1250) — matrix operand order repaired and checked by reading, not compiled, not run on hardware
//
#include <hip/hip_runtime.h>

typedef __attribute__((ext_vector_type(16))) _Float16 v16h;
typedef __attribute__((ext_vector_type(8)))  float    v8f;

#define DIMC 512
#define LTOT 1024
#define NB   64
#define KCL  64
#define KP   80          // 65 cluster rows padded to 5 WMMA tiles
#define EPS  1e-12f

// ---- workspace layout (bytes, 256-aligned) ----
static constexpr size_t OFF_WA1   = 0;                              // conv_w A-frags f16: 5*16*512*2      = 81,920
static constexpr size_t OFF_B1    = 81920;                          // xn B-frags GEMM1: 64*64*16*512*2    = 67,108,864
static constexpr size_t OFF_B2    = OFF_B1 + 67108864ull;           // xnT B-frags GEMM2: 64*32*32*512*2   = 67,108,864
static constexpr size_t OFF_RNORM = OFF_B2 + 67108864ull;           // f32 64*1024                         = 262,144
static constexpr size_t OFF_AFRAG = OFF_RNORM + 262144ull;          // a A-frags f16: 64*5*32*512*2        = 10,485,760
static constexpr size_t OFF_ASUM  = OFF_AFRAG + 10485760ull;        // f32 64*80                           = 20,480
static constexpr size_t OFF_NSUM  = OFF_ASUM + 20480ull;            // f32 64                              = 256
static constexpr size_t OFF_W     = OFF_NSUM + 256ull;              // weighted f32 64*80*512              = 10,485,760

__global__ void k_init(float* asum, float* nsum) {
  int i = blockIdx.x * blockDim.x + threadIdx.x;
  if (i < NB * KP) asum[i] = 0.f;
  if (i < NB)      nsum[i] = 0.f;
}

// conv_w (65x512) -> f16 A-fragments, rows >=65 zero
__global__ void k_wfrag(const float* __restrict__ conv_w, _Float16* __restrict__ wA) {
  int frag = blockIdx.x;               // 0..79  (mt*16 + cs)
  int lane = threadIdx.x;              // 0..31
  int mt = frag >> 4, cs = frag & 15;
  int m   = mt * 16 + (lane & 15);
  int khi = lane >> 4;
  v16h o;
#pragma unroll
  for (int h = 0; h < 16; ++h) {
    int kk = khi * 16 + h;                           // A-layout K index
    int c  = cs * 32 + kk;
    o[h] = (_Float16)((m < 65) ? conv_w[m * DIMC + c] : 0.f);
  }
  *(v16h*)(wA + (size_t)frag * 512 + lane * 16) = o;
}

// per-pixel inverse L2 norm over channels
__global__ void k_rnorm(const float* __restrict__ x, float* __restrict__ rnorm) {
  int idx = blockIdx.x * blockDim.x + threadIdx.x;   // n*1024 + l
  int n = idx >> 10, l = idx & 1023;
  const float* xp = x + (size_t)n * DIMC * LTOT + l;
  float s = 0.f;
  for (int c = 0; c < DIMC; ++c) { float v = xp[(size_t)c * LTOT]; s += v * v; }
  rnorm[idx] = 1.f / fmaxf(sqrtf(s), EPS);
}

// normalized x -> f16 B-fragments for GEMM1 (K = channel dim)
__global__ void k_b1(const float* __restrict__ x, const float* __restrict__ rnorm,
                     _Float16* __restrict__ b1) {
  int wave = (blockIdx.x * blockDim.x + threadIdx.x) >> 5;   // frag id = (n*64+lt)*16+cs
  int lane = threadIdx.x & 31;
  int cs = wave & 15; int t = wave >> 4; int lt = t & 63; int n = t >> 6;
  int l   = lt * 16 + (lane & 15);
  int khi = lane >> 4;
  float sc = rnorm[n * LTOT + l];
  const float* xp = x + (size_t)n * DIMC * LTOT + l;
  v16h o;
#pragma unroll
  for (int h = 0; h < 16; ++h) {
    int c = cs * 32 + khi * 16 + h;                 // B-layout K index
    o[h] = (_Float16)(xp[(size_t)c * LTOT] * sc);
  }
  *(v16h*)(b1 + (size_t)wave * 512 + lane * 16) = o;
}

// normalized x^T -> f16 B-fragments for GEMM2 (K = pixel dim)
__global__ void k_b2(const float* __restrict__ x, const float* __restrict__ rnorm,
                     _Float16* __restrict__ b2) {
  int wave = (blockIdx.x * blockDim.x + threadIdx.x) >> 5;   // frag id = (n*32+ls)*32+ct
  int lane = threadIdx.x & 31;
  int ct = wave & 31; int t = wave >> 5; int ls = t & 31; int n = t >> 5;
  int c     = ct * 16 + (lane & 15);
  int khi   = lane >> 4;
  int lbase = ls * 32 + khi * 16;
  const float* xp = x + ((size_t)(n * DIMC + c)) * LTOT + lbase;
  const float* rp = rnorm + n * LTOT + lbase;
  v16h o;
#pragma unroll
  for (int h = 0; h < 16; ++h) o[h] = (_Float16)(xp[h] * rp[h]);
  *(v16h*)(b2 + (size_t)wave * 512 + lane * 16) = o;
}

// GEMM1 (logits) + bias + column softmax + a_sum + write a as GEMM2 A-fragments
__global__ void __launch_bounds__(256)
k_gemm1(const _Float16* __restrict__ wA, const _Float16* __restrict__ b1,
        const float* __restrict__ conv_b, _Float16* __restrict__ aFrag,
        float* __restrict__ asum) {
  __shared__ _Float16 alds[KP * 136];                // padded stride kills bank conflicts
  int n = blockIdx.x >> 3, lb = blockIdx.x & 7;      // 8 pixel-blocks of 128 per image
  int wave = threadIdx.x >> 5, lane = threadIdx.x & 31;
  int lt = lb * 8 + wave;
  v8f acc[5] = {};
  const _Float16* bbase = b1 + ((size_t)(n * 64 + lt) * 16) * 512 + lane * 16;
#pragma unroll
  for (int cs = 0; cs < 16; ++cs) {
    if (cs < 15) __builtin_prefetch(bbase + (size_t)(cs + 1) * 512, 0, 1);
    v16h B = *(const v16h*)(bbase + (size_t)cs * 512);
#pragma unroll
    for (int mt = 0; mt < 5; ++mt) {
      v16h A = *(const v16h*)(wA + (size_t)(mt * 16 + cs) * 512 + lane * 16);
      acc[mt] = __builtin_amdgcn_wmma_f32_16x16x32_f16(false, A, false, B,
                                                       (short)0, acc[mt], false, false);
    }
  }
  int hi = lane >> 4, col = lane & 15;
  float lg[5][8];
  float mx = -3.4e38f;
#pragma unroll
  for (int mt = 0; mt < 5; ++mt)
#pragma unroll
    for (int j = 0; j < 8; ++j) {
      int m = mt * 16 + j + 8 * hi;
      float v = (m < 65) ? (acc[mt][j] + conv_b[m]) : -3.4e38f;
      lg[mt][j] = v; mx = fmaxf(mx, v);
    }
  mx = fmaxf(mx, __shfl_xor(mx, 16, 32));            // column spans lane pair (xor 16)
  float sum = 0.f;
#pragma unroll
  for (int mt = 0; mt < 5; ++mt)
#pragma unroll
    for (int j = 0; j < 8; ++j) {
      int m = mt * 16 + j + 8 * hi;
      float e = (m < 65) ? __expf(lg[mt][j] - mx) : 0.f;
      lg[mt][j] = e; sum += e;
    }
  sum += __shfl_xor(sum, 16, 32);
  float inv = 1.f / sum;
#pragma unroll
  for (int mt = 0; mt < 5; ++mt)
#pragma unroll
    for (int j = 0; j < 8; ++j) {
      int m = mt * 16 + j + 8 * hi;
      alds[m * 136 + wave * 16 + col] = (_Float16)(lg[mt][j] * inv);
    }
  __syncthreads();
  if (threadIdx.x < KP) {                            // a_sum over this 128-pixel block
    int m = threadIdx.x; float s = 0.f;
    for (int ll = 0; ll < 128; ++ll) s += (float)alds[m * 136 + ll];
    atomicAdd(&asum[n * KP + m], s);
  }
  for (int f = wave; f < 20; f += 8) {               // 5 mt * 4 local K-steps
    int mt = f >> 2, lsl = f & 3;
    int m = mt * 16 + (lane & 15);
    int khi = lane >> 4;
    v16h o;
#pragma unroll
    for (int h = 0; h < 16; ++h) {
      int klocal = khi * 16 + h;
      o[h] = alds[m * 136 + lsl * 32 + klocal];
    }
    *(v16h*)(aFrag + (size_t)((n * 5 + mt) * 32 + lb * 4 + lsl) * 512 + lane * 16) = o;
  }
}

// GEMM2: weighted(80x512) = a(80x1024) x xn^T(1024x512) per image
__global__ void __launch_bounds__(256)
k_gemm2(const _Float16* __restrict__ aFrag, const _Float16* __restrict__ b2,
        float* __restrict__ wout) {
  int n = blockIdx.x >> 2, cb = blockIdx.x & 3;      // 4 channel slices of 128
  int wave = threadIdx.x >> 5, lane = threadIdx.x & 31;
  int ct = cb * 8 + wave;
  v8f acc[5] = {};
  const _Float16* bbase = b2 + ((size_t)(n * 32) * 32 + ct) * 512 + lane * 16;
#pragma unroll 4
  for (int ls = 0; ls < 32; ++ls) {
    if (ls < 31) __builtin_prefetch(bbase + (size_t)(ls + 1) * 32 * 512, 0, 1);
    v16h B = *(const v16h*)(bbase + (size_t)ls * 32 * 512);
#pragma unroll
    for (int mt = 0; mt < 5; ++mt) {
      v16h A = *(const v16h*)(aFrag + (size_t)((n * 5 + mt) * 32 + ls) * 512 + lane * 16);
      acc[mt] = __builtin_amdgcn_wmma_f32_16x16x32_f16(false, A, false, B,
                                                       (short)0, acc[mt], false, false);
    }
  }
  int hi = lane >> 4, col = lane & 15;
  int c = ct * 16 + col;
#pragma unroll
  for (int mt = 0; mt < 5; ++mt)
#pragma unroll
    for (int j = 0; j < 8; ++j) {
      int m = mt * 16 + j + 8 * hi;
      wout[(size_t)(n * KP + m) * DIMC + c] = acc[mt][j];
    }
}

// residual vs centroids + per-row L2 normalize + per-image sumsq accumulation
__global__ void k_vlad(const float* __restrict__ wout, const float* __restrict__ asum,
                       const float* __restrict__ centroids, float* __restrict__ out,
                       float* __restrict__ nsum) {
  __shared__ float red[256];
  int n = blockIdx.x >> 6, k = blockIdx.x & 63;
  int tid = threadIdx.x;
  float as = asum[n * KP + k];
  const float* wp = wout + (size_t)(n * KP + k) * DIMC;
  const float* cp = centroids + (size_t)k * DIMC;
  float v0 = wp[tid]       - as * cp[tid];
  float v1 = wp[tid + 256] - as * cp[tid + 256];
  red[tid] = v0 * v0 + v1 * v1;
  __syncthreads();
  for (int s = 128; s > 0; s >>= 1) {
    if (tid < s) red[tid] += red[tid + s];
    __syncthreads();
  }
  float ss = red[0];
  float sc = 1.f / fmaxf(sqrtf(ss), EPS);
  float* op = out + (size_t)n * (KCL * DIMC) + (size_t)k * DIMC;
  op[tid]       = v0 * sc;
  op[tid + 256] = v1 * sc;
  if (tid == 0) atomicAdd(&nsum[n], ss * sc * sc);
}

// final per-image L2 normalize
__global__ void k_final(float* __restrict__ out, const float* __restrict__ nsum) {
  size_t i = (size_t)blockIdx.x * blockDim.x + threadIdx.x;
  int n = (int)(i >> 15);
  out[i] *= 1.f / fmaxf(sqrtf(nsum[n]), EPS);
}

extern "C" void kernel_launch(void* const* d_in, const int* in_sizes, int n_in,
                              void* d_out, int out_size, void* d_ws, size_t ws_size,
                              hipStream_t stream) {
  const float* x         = (const float*)d_in[0];   // (64,512,32,32)
  const float* centroids = (const float*)d_in[1];   // (65,512)
  const float* conv_w    = (const float*)d_in[2];   // (65,512)
  const float* conv_b    = (const float*)d_in[3];   // (65,)
  float* out = (float*)d_out;

  char* ws = (char*)d_ws;
  _Float16* wA    = (_Float16*)(ws + OFF_WA1);
  _Float16* b1    = (_Float16*)(ws + OFF_B1);
  _Float16* b2    = (_Float16*)(ws + OFF_B2);
  float*    rnorm = (float*)   (ws + OFF_RNORM);
  _Float16* aFrag = (_Float16*)(ws + OFF_AFRAG);
  float*    asum  = (float*)   (ws + OFF_ASUM);
  float*    nsum  = (float*)   (ws + OFF_NSUM);
  float*    wout  = (float*)   (ws + OFF_W);

  k_init <<<20,   256, 0, stream>>>(asum, nsum);
  k_wfrag<<<80,    32, 0, stream>>>(conv_w, wA);
  k_rnorm<<<256,  256, 0, stream>>>(x, rnorm);
  k_b1   <<<8192, 256, 0, stream>>>(x, rnorm, b1);
  k_b2   <<<8192, 256, 0, stream>>>(x, rnorm, b2);
  k_gemm1<<<512,  256, 0, stream>>>(wA, b1, conv_b, aFrag, asum);
  k_gemm2<<<256,  256, 0, stream>>>(aFrag, b2, wout);
  k_vlad <<<4096, 256, 0, stream>>>(wout, asum, centroids, out, nsum);
  k_final<<<8192, 256, 0, stream>>>(out, nsum);
}
